// EGNNLayer_87643102642597
// MI455X (gfx1250) — matrix operand order repaired and checked by reading, not compiled, not run on hardware
//
#include <hip/hip_runtime.h>
#include <hip/hip_bf16.h>

#define N_NODE 4096
#define KNN 32
#define HIDDEN 128
#define RMS_EPS 1e-5f
#define NEDGE (N_NODE * KNN)   // 131072

typedef _Float16 v16h __attribute__((ext_vector_type(16)));
typedef _Float16 v8h  __attribute__((ext_vector_type(8)));
typedef float    v8f  __attribute__((ext_vector_type(8)));

// Fast SiLU: v_exp_f32 + v_rcp_f32 (no IEEE div chain).
__device__ __forceinline__ float silu_fast(float x) {
    return x * __builtin_amdgcn_rcpf(1.0f + __expf(-x));
}

// ---------------------------------------------------------------------------
// Kernel 1: kNN — one wave32 per node, distances staged in LDS, 32 rounds of
// cooperative min-selection (tie-break on lower index to match lax.top_k).
// ---------------------------------------------------------------------------
__global__ __launch_bounds__(64) void knn_kernel(const float* __restrict__ pos,
                                                 int* __restrict__ nbr) {
    __shared__ float sdist[2 * N_NODE];          // 32 KB: 2 waves per block
    const int lane = threadIdx.x & 31;
    const int w    = threadIdx.x >> 5;
    const int node = (blockIdx.x << 1) + w;
    float* dist = &sdist[w * N_NODE];

    const float px = pos[node * 3 + 0];
    const float py = pos[node * 3 + 1];
    const float pz = pos[node * 3 + 2];

    for (int j = lane; j < N_NODE; j += 32) {
        float dx = px - pos[j * 3 + 0];
        float dy = py - pos[j * 3 + 1];
        float dz = pz - pos[j * 3 + 2];
        float d  = dx * dx + dy * dy + dz * dz;
        if (j == node) d = __builtin_inff();
        dist[j] = d;
    }
    __syncthreads();

    for (int kk = 0; kk < KNN; ++kk) {
        float best  = __builtin_inff();
        int   bestj = 0x7fffffff;
        for (int j = lane; j < N_NODE; j += 32) {
            float d = dist[j];
            if (d < best || (d == best && j < bestj)) { best = d; bestj = j; }
        }
        #pragma unroll
        for (int m = 16; m >= 1; m >>= 1) {
            float ob = __shfl_xor(best, m, 32);
            int   oj = __shfl_xor(bestj, m, 32);
            if (ob < best || (ob == best && oj < bestj)) { best = ob; bestj = oj; }
        }
        if (lane == 0) {
            nbr[node * KNN + kk] = bestj;
            dist[bestj] = __builtin_inff();
        }
        __syncthreads();
    }
}

// ---------------------------------------------------------------------------
// Kernel 2: fused edge MLP. One wave per 16-edge tile. Layer-2 (128x128 GEMM)
// runs on v_wmma_f32_16x16x32_f16 with f32 accumulation. A fragments are built
// in-register per the CDNA5 16-bit A layout; B (W2^T) is f16 in LDS.
// ---------------------------------------------------------------------------
__global__ __launch_bounds__(256) void edge_kernel(
        const float* __restrict__ pos,  const float* __restrict__ tptr,
        const float* __restrict__ W1,   const float* __restrict__ b1,
        const float* __restrict__ g1,   const float* __restrict__ W2,
        const float* __restrict__ b2,   const float* __restrict__ g2,
        const float* __restrict__ W3,   const float* __restrict__ b3,
        const int*   __restrict__ nbr,  float* __restrict__ trans) {
    constexpr int W2S = HIDDEN + 8;                       // pad: break LDS bank conflicts
    __shared__ __attribute__((aligned(16))) _Float16 w2h[HIDDEN * W2S]; // ~34 KB
    __shared__ float escal[8][16];

    const int tid = threadIdx.x;
    // Cooperative W2 f32 -> f16 conversion into LDS (row-major: [c][k]).
    for (int idx = tid; idx < HIDDEN * HIDDEN; idx += 256) {
        int rr = idx >> 7, cc = idx & 127;
        w2h[rr * W2S + cc] = (_Float16)W2[idx];
    }
    __syncthreads();

    const int lane = tid & 31;
    const int wv   = tid >> 5;
    const int tile = (blockIdx.x << 3) + wv;   // 16-edge tile index
    const int m    = lane & 15;                // A-matrix row (edge within tile)
    const int hi   = lane >> 4;

    const int e = tile * 16 + m;
    const int s = e >> 5;                      // sender node
    const int r = nbr[e];                      // receiver node
    const float t = tptr[0];

    const float cdx = pos[s * 3 + 0] - pos[r * 3 + 0];
    const float cdy = pos[s * 3 + 1] - pos[r * 3 + 1];
    const float cdz = pos[s * 3 + 2] - pos[r * 3 + 2];
    const float radial = cdx * cdx + cdy * cdy + cdz * cdz;

    // ---- layer 1: this lane's 64 channels (the ones its A-fragment needs) ----
    float h1v[4][16];
    float ss = 0.f;
    #pragma unroll
    for (int kk = 0; kk < 4; ++kk)
        #pragma unroll
        for (int g = 0; g < 2; ++g)
            #pragma unroll
            for (int rr = 0; rr < 8; ++rr) {
                const int k = kk * 32 + g * 16 + hi * 8 + rr;
                const float v = W1[2 * k] * radial + W1[2 * k + 1] * t + b1[k];
                h1v[kk][g * 8 + rr] = v;
                ss += v * v;
            }
    ss += __shfl_xor(ss, 16, 32);              // pair lane covers the other 64 ch.
    const float rn1 = rsqrtf(ss * (1.0f / HIDDEN) + RMS_EPS);

    v16h afrag[4];
    #pragma unroll
    for (int kk = 0; kk < 4; ++kk)
        #pragma unroll
        for (int g = 0; g < 2; ++g)
            #pragma unroll
            for (int rr = 0; rr < 8; ++rr) {
                const int k = kk * 32 + g * 16 + hi * 8 + rr;
                const float x = silu_fast(h1v[kk][g * 8 + rr] * rn1 * g1[k]);
                afrag[kk][g * 8 + rr] = (_Float16)x;
            }

    // ---- layer 2: 16x128 @ 128x128 via 8 N-tiles x 4 K-steps of WMMA ----
    v8f cacc[8];
    #pragma unroll
    for (int nt = 0; nt < 8; ++nt) {
        v8f acc = {};
        const int col = nt * 16 + m;           // B column n (W2 row)
        #pragma unroll
        for (int kk = 0; kk < 4; ++kk) {
            const _Float16* bp = &w2h[col * W2S + kk * 32 + hi * 8];
            v8h blo = *(const v8h*)bp;
            v8h bhi = *(const v8h*)(bp + 16);
            v16h bfrag = __builtin_shufflevector(blo, bhi,
                0, 1, 2, 3, 4, 5, 6, 7, 8, 9, 10, 11, 12, 13, 14, 15);
            acc = __builtin_amdgcn_wmma_f32_16x16x32_f16(
                false, afrag[kk], false, bfrag, (short)0, acc, false, false);
        }
        cacc[nt] = acc;
    }

    // ---- bias + RMSNorm stats in native C layout (row M = hi*8+v) ----
    float rowss[8];
    #pragma unroll
    for (int v = 0; v < 8; ++v) rowss[v] = 0.f;
    #pragma unroll
    for (int nt = 0; nt < 8; ++nt) {
        const float bb = b2[nt * 16 + m];
        #pragma unroll
        for (int v = 0; v < 8; ++v) {
            const float x = cacc[nt][v] + bb;
            cacc[nt][v] = x;
            rowss[v] += x * x;
        }
    }
    #pragma unroll
    for (int v = 0; v < 8; ++v) {
        #pragma unroll
        for (int msk = 8; msk >= 1; msk >>= 1)
            rowss[v] += __shfl_xor(rowss[v], msk, 32);
    }
    float rnorm[8];
    #pragma unroll
    for (int v = 0; v < 8; ++v)
        rnorm[v] = rsqrtf(rowss[v] * (1.0f / HIDDEN) + RMS_EPS);

    // ---- SiLU + layer-3 dot with W3 ----
    float pacc[8];
    #pragma unroll
    for (int v = 0; v < 8; ++v) pacc[v] = 0.f;
    #pragma unroll
    for (int nt = 0; nt < 8; ++nt) {
        const float gg = g2[nt * 16 + m];
        const float ww = W3[nt * 16 + m];
        #pragma unroll
        for (int v = 0; v < 8; ++v) {
            const float x = silu_fast(cacc[nt][v] * rnorm[v] * gg);
            pacc[v] += x * ww;
        }
    }
    #pragma unroll
    for (int v = 0; v < 8; ++v) {
        #pragma unroll
        for (int msk = 8; msk >= 1; msk >>= 1)
            pacc[v] += __shfl_xor(pacc[v], msk, 32);
    }

    const float bb3 = b3[0];
    if (m == 0) {                               // 2 lanes publish the 16 edge scalars
        #pragma unroll
        for (int v = 0; v < 8; ++v)
            escal[wv][hi * 8 + v] = pacc[v] + bb3;
    }
    __syncthreads();

    if (lane < 16) {                            // lane == its own edge m
        const float sc = escal[wv][lane];
        float* o = &trans[(size_t)e * 3];
        o[0] = cdx * sc;
        o[1] = cdy * sc;
        o[2] = cdz * sc;
    }
}

// ---------------------------------------------------------------------------
// Kernel 3: contiguous 32-edge segment sum per node (no atomics), + pos.
// ---------------------------------------------------------------------------
__global__ __launch_bounds__(256) void reduce_kernel(const float* __restrict__ pos,
                                                     const float* __restrict__ trans,
                                                     float* __restrict__ out) {
    const int tid = blockIdx.x * blockDim.x + threadIdx.x;
    if (tid >= N_NODE * 3) return;
    const int i = tid / 3;
    const int c = tid % 3;
    float ssum = 0.f;
    const float* base = &trans[(size_t)i * KNN * 3 + c];
    #pragma unroll 8
    for (int j = 0; j < KNN; ++j) ssum += base[j * 3];
    out[tid] = pos[tid] + ssum * (1.0f / KNN);
}

// ---------------------------------------------------------------------------
extern "C" void kernel_launch(void* const* d_in, const int* in_sizes, int n_in,
                              void* d_out, int out_size, void* d_ws, size_t ws_size,
                              hipStream_t stream) {
    (void)in_sizes; (void)n_in; (void)out_size; (void)ws_size;
    const float* pos = (const float*)d_in[0];
    const float* t   = (const float*)d_in[1];
    const float* W1  = (const float*)d_in[2];
    const float* b1  = (const float*)d_in[3];
    const float* g1  = (const float*)d_in[4];
    const float* W2  = (const float*)d_in[5];
    const float* b2  = (const float*)d_in[6];
    const float* g2  = (const float*)d_in[7];
    const float* W3  = (const float*)d_in[8];
    const float* b3  = (const float*)d_in[9];
    float* out = (float*)d_out;

    int*   nbr   = (int*)d_ws;                                          // 512 KB
    float* trans = (float*)((char*)d_ws + (size_t)NEDGE * sizeof(int)); // 1.5 MB

    knn_kernel<<<N_NODE / 2, 64, 0, stream>>>(pos, nbr);
    edge_kernel<<<NEDGE / 128, 256, 0, stream>>>(pos, t, W1, b1, g1, W2, b2, g2,
                                                 W3, b3, nbr, trans);
    reduce_kernel<<<(N_NODE * 3 + 255) / 256, 256, 0, stream>>>(pos, trans, out);
}
